// ContrastiveLoss_65944927863068
// MI455X (gfx1250) — matrix operand order repaired and checked by reading, not compiled, not run on hardware
//
#include <hip/hip_runtime.h>
#include <math.h>

typedef __attribute__((ext_vector_type(16))) _Float16 v16h;
typedef __attribute__((ext_vector_type(8)))  float    v8f;
typedef __attribute__((ext_vector_type(4)))  int      v4i;

#define NI 128   // captions
#define NJ 128   // images
#define RR 36    // regions
#define WW 32    // words
#define DD 1024  // feature dim

#define AST 136  // LDS half-stride (128 + 8 pad -> 4-bank rotation, 16B aligned rows)
#define KC  128  // K-chunk (double buffered)
#define NCHUNK (DD / KC)

#if __has_builtin(__builtin_amdgcn_global_load_async_to_lds_b128)
#define HAVE_ASYNC 1
#else
#define HAVE_ASYNC 0
#endif

union V16 { v16h v; uint4 u[2]; };

// 16B global -> LDS copy; async (no VGPR round trip, ASYNCcnt tracked) when available
__device__ static inline void cp16_g2l(void* lds, const void* g) {
#if HAVE_ASYNC
    __builtin_amdgcn_global_load_async_to_lds_b128(
        (__attribute__((address_space(1))) v4i*)(void*)g,
        (__attribute__((address_space(3))) v4i*)lds,
        0, 0);
#else
    *(uint4*)lds = *(const uint4*)g;
#endif
}

__device__ static inline void wait_async_copies() {
#if HAVE_ASYNC
#if __has_builtin(__builtin_amdgcn_s_wait_asynccnt)
    __builtin_amdgcn_s_wait_asynccnt(0);
#else
    asm volatile("s_wait_asynccnt 0x0" ::: "memory");
#endif
#endif
}

// ---------------- fp32 -> f16 conversion ----------------
__global__ void cvt_f32_f16(const float* __restrict__ src, _Float16* __restrict__ dst, int n) {
    int idx = blockIdx.x * blockDim.x + threadIdx.x;
    if (idx < n) dst[idx] = (_Float16)src[idx];
}

// ---------------- per-image Gram matrices G_j = im_j im_j^T (fp32) ----------------
__global__ void gram_kernel(const float* __restrict__ im, float* __restrict__ G) {
    int j = blockIdx.x;
    const float* base = im + (size_t)j * RR * DD;
    for (int p = threadIdx.x; p < RR * RR; p += blockDim.x) {
        int r = p / RR, rp = p % RR;
        const float4* a = (const float4*)(base + r * DD);
        const float4* b = (const float4*)(base + rp * DD);
        float acc = 0.f;
        for (int d = 0; d < DD / 4; d++) {
            float4 av = a[d], bv = b[d];
            acc = fmaf(av.x, bv.x, acc);
            acc = fmaf(av.y, bv.y, acc);
            acc = fmaf(av.z, bv.z, acc);
            acc = fmaf(av.w, bv.w, acc);
        }
        G[(size_t)j * RR * RR + p] = acc;
    }
}

// ---------------- exact word norms ||s[i,w]|| ----------------
__global__ void snorm_kernel(const float* __restrict__ s, float* __restrict__ sn) {
    int i = blockIdx.x;
    int t = threadIdx.x;
    int w = t >> 5, lane = t & 31;
    const float* p = s + ((size_t)i * WW + w) * DD;
    float acc = 0.f;
    for (int d = lane; d < DD; d += 32) { float v = p[d]; acc = fmaf(v, v, acc); }
    for (int off = 16; off > 0; off >>= 1) acc += __shfl_xor(acc, off, 32);
    if (lane == 0) sn[i * WW + w] = sqrtf(acc);
}

// ---------------- main fused cross-attention kernel: one WG per (i,j) ----------------
__global__ __launch_bounds__(192) void xattn_kernel(
    const _Float16* __restrict__ imh, const _Float16* __restrict__ sh,
    const float* __restrict__ G, const float* __restrict__ sn,
    float* __restrict__ scores) {

    const int i = blockIdx.x;   // caption
    const int j = blockIdx.y;   // image
    const int t = threadIdx.x;
    const int lane = t & 31;
    const int wave = t >> 5;    // 0..5
    const int mt = wave >> 1;   // 0..2 : region tile (rows mt*16..mt*16+15, padded to 48)
    const int nt = wave & 1;    // 0..1 : word tile
    const int hg = lane >> 4;   // half-wave group
    const int l15 = lane & 15;

    __shared__ _Float16 AshBuf[2][48 * AST];
    __shared__ _Float16 BshBuf[2][WW * AST];
    __shared__ float rawS[RR * 33];
    __shared__ float tS[RR * 33];
    __shared__ float pNum[WW * 6];
    __shared__ float pQuad[WW * 6];
    __shared__ float simS[WW];

    v8f acc;
    for (int q = 0; q < 8; q++) acc[q] = 0.f;

    const uint4* gA = (const uint4*)(imh + (size_t)j * RR * DD);  // row stride DD/8 = 128 uint4
    const uint4* gB = (const uint4*)(sh  + (size_t)i * WW * DD);

    // zero-fill pad rows 36..47 of both A buffers once (async copies never touch them)
    {
        uint4 z = make_uint4(0u, 0u, 0u, 0u);
        int r = RR + (t >> 4);  // 36..47
        int c = t & 15;
        *(uint4*)&AshBuf[0][r * AST + c * 8] = z;
        *(uint4*)&AshBuf[1][r * AST + c * 8] = z;
    }

    // prefetch chunk 0 into buffer 0
    {
        for (int idx = t; idx < RR * (KC / 8); idx += 192) {
            int r = idx >> 4, c = idx & 15;
            cp16_g2l(&AshBuf[0][r * AST + c * 8], &gA[r * 128 + c]);
        }
        for (int idx = t; idx < WW * (KC / 8); idx += 192) {
            int r = idx >> 4, c = idx & 15;
            cp16_g2l(&BshBuf[0][r * AST + c * 8], &gB[r * 128 + c]);
        }
    }

    const int am = mt * 16 + l15;   // A row (M)
    const int bw = nt * 16 + l15;   // B col (N) -> word row in Bsh

    for (int ch = 0; ch < NCHUNK; ch++) {
        wait_async_copies();     // my chunk-ch copies are in LDS
        __syncthreads();         // everyone's copies are in LDS; prev compute finished

        // prefetch chunk ch+1 into the other buffer while we compute on this one
        if (ch + 1 < NCHUNK) {
            int nb = (ch + 1) & 1;
            int kq = ((ch + 1) * KC) >> 3;  // global column offset in uint4
            for (int idx = t; idx < RR * (KC / 8); idx += 192) {
                int r = idx >> 4, c = idx & 15;
                cp16_g2l(&AshBuf[nb][r * AST + c * 8], &gA[r * 128 + kq + c]);
            }
            for (int idx = t; idx < WW * (KC / 8); idx += 192) {
                int r = idx >> 4, c = idx & 15;
                cp16_g2l(&BshBuf[nb][r * AST + c * 8], &gB[r * 128 + kq + c]);
            }
        }

        const _Float16* Abuf = AshBuf[ch & 1];
        const _Float16* Bbuf = BshBuf[ch & 1];
#pragma unroll
        for (int kk = 0; kk < KC; kk += 32) {
            V16 A, B;
            // A 16x32 f16 layout: lane<16 holds K 0..7,16..23 ; lane>=16 holds K 8..15,24..31
            int ak = kk + hg * 8;
            A.u[0] = *(const uint4*)&Abuf[am * AST + ak];
            A.u[1] = *(const uint4*)&Abuf[am * AST + ak + 16];
            // B 32x16 f16 layout: lanes 0-15 hold K 0..7,16..23 ; lanes 16-31 hold K 8..15,24..31
            int bk = kk + hg * 8;
            B.u[0] = *(const uint4*)&Bbuf[bw * AST + bk];
            B.u[1] = *(const uint4*)&Bbuf[bw * AST + bk + 16];
            acc = __builtin_amdgcn_wmma_f32_16x16x32_f16(
                false, A.v, false, B.v, (short)0, acc, false, false);
        }
        __syncthreads();         // done reading this buffer; next iter may overwrite it
    }

    // scatter C tile to LDS: raw[r][w] ; C layout: vgpr v -> M = v + 8*(lane>=16), N = lane&15
    {
        int w = nt * 16 + l15;
        for (int v = 0; v < 8; v++) {
            int r = mt * 16 + hg * 8 + v;
            if (r < RR) rawS[r * 33 + w] = acc[v];
        }
    }
    __syncthreads();

    // Step A: LeakyReLU(0.1) + L2-normalize over words, per region row
    if (t < RR) {
        float ss = 0.f;
        for (int w = 0; w < WW; w++) {
            float v = rawS[t * 33 + w];
            float lv = v > 0.f ? v : 0.1f * v;
            tS[t * 33 + w] = lv;
            ss = fmaf(lv, lv, ss);
        }
        float inv = 1.f / (sqrtf(ss) + 1e-8f);
        for (int w = 0; w < WW; w++) tS[t * 33 + w] *= inv;
    }
    __syncthreads();

    // Step B: softmax over regions (lambda=9) per word; T[w,r] kept at tS[r][w]
    if (t < WW) {
        float m = -1e30f;
        for (int r = 0; r < RR; r++) m = fmaxf(m, 9.0f * tS[r * 33 + t]);
        float sum = 0.f;
        for (int r = 0; r < RR; r++) {
            float e = __expf(9.0f * tS[r * 33 + t] - m);
            tS[r * 33 + t] = e;
            sum += e;
        }
        float invs = 1.f / sum;
        for (int r = 0; r < RR; r++) tS[r * 33 + t] *= invs;
    }
    __syncthreads();

    // Step C: num[w] = sum_r T[w,r] raw[r,w] ; ||wctx||^2 = T^T G_j T  (6 threads/word)
    {
        int w = t / 6, p = t % 6;
        const float* gj = G + (size_t)j * RR * RR;
        float num = 0.f, quad = 0.f;
        for (int r = p * 6; r < p * 6 + 6; r++) {
            float tw = tS[r * 33 + w];
            num = fmaf(tw, rawS[r * 33 + w], num);
            float g = 0.f;
            for (int rp = 0; rp < RR; rp++) g = fmaf(gj[r * RR + rp], tS[rp * 33 + w], g);
            quad = fmaf(tw, g, quad);
        }
        pNum[w * 6 + p] = num;
        pQuad[w * 6 + p] = quad;
    }
    __syncthreads();

    if (t < WW) {
        float num = 0.f, quad = 0.f;
        for (int p = 0; p < 6; p++) { num += pNum[t * 6 + p]; quad += pQuad[t * 6 + p]; }
        float w2 = sqrtf(fmaxf(quad, 0.f));
        float denom = fmaxf(sn[i * WW + t] * w2, 1e-8f);
        simS[t] = num / denom;
    }
    __syncthreads();

    // LSE over words (lambda=6) -> scores[j,i] (row.T)
    if (t == 0) {
        float m = -1e30f;
        for (int w = 0; w < WW; w++) m = fmaxf(m, 6.0f * simS[w]);
        float sum = 0.f;
        for (int w = 0; w < WW; w++) sum += __expf(6.0f * simS[w] - m);
        scores[(size_t)j * NI + i] = (m + __logf(sum)) / 6.0f;
    }
}

// ---------------- hinge loss with max_violation ----------------
__global__ void loss_kernel(const float* __restrict__ S, float* __restrict__ out) {
    int t = threadIdx.x;  // 0..127
    __shared__ float diag[NJ];
    __shared__ float red[NJ];
    diag[t] = S[t * NI + t];
    __syncthreads();
    float ms = 0.f;  // max over i of cost_s[t, i]
    for (int i = 0; i < NI; i++)
        if (i != t) ms = fmaxf(ms, fmaxf(0.f, 0.2f + S[t * NI + i] - diag[t]));
    float mi = 0.f;  // max over j of cost_im[j, t]
    for (int j = 0; j < NJ; j++)
        if (j != t) mi = fmaxf(mi, fmaxf(0.f, 0.2f + S[j * NI + t] - diag[t]));
    red[t] = ms + mi;
    __syncthreads();
    for (int off = 64; off > 0; off >>= 1) {
        if (t < off) red[t] += red[t + off];
        __syncthreads();
    }
    if (t == 0) out[0] = red[0];
}

extern "C" void kernel_launch(void* const* d_in, const int* in_sizes, int n_in,
                              void* d_out, int out_size, void* d_ws, size_t ws_size,
                              hipStream_t stream) {
    const float* im = (const float*)d_in[0];  // (128,36,1024)
    const float* s  = (const float*)d_in[1];  // (128,32,1024)
    float* out = (float*)d_out;

    char* ws = (char*)d_ws;
    const size_t n_im = (size_t)NJ * RR * DD;  // 4,718,592
    const size_t n_s  = (size_t)NI * WW * DD;  // 4,194,304
    _Float16* imh = (_Float16*)(ws);
    _Float16* shp = (_Float16*)(ws + n_im * 2);
    float* G   = (float*)(ws + n_im * 2 + n_s * 2);
    float* sn  = (float*)(ws + n_im * 2 + n_s * 2 + (size_t)NJ * RR * RR * 4);
    float* sc  = (float*)(ws + n_im * 2 + n_s * 2 + (size_t)NJ * RR * RR * 4 + (size_t)NI * WW * 4);

    cvt_f32_f16<<<(int)((n_im + 255) / 256), 256, 0, stream>>>(im, imh, (int)n_im);
    cvt_f32_f16<<<(int)((n_s + 255) / 256), 256, 0, stream>>>(s, shp, (int)n_s);
    gram_kernel<<<NJ, 256, 0, stream>>>(im, G);
    snorm_kernel<<<NI, 1024, 0, stream>>>(s, sn);

    dim3 grid(NI, NJ);
    xattn_kernel<<<grid, 192, 0, stream>>>(imh, shp, G, sn, sc);

    loss_kernel<<<1, 128, 0, stream>>>(sc, out);
}
